// MOELayer_14869176779392
// MI455X (gfx1250) — matrix operand order repaired and checked by reading, not compiled, not run on hardware
//
#include <hip/hip_runtime.h>

#define E_NUM 8
#define D_DIM 512
#define H_DIM 2048
#define N_TOK 8192
#define TILE_M 32
#define STRIDE_A 520   // 512 + 8 bf16 pad -> 4-bank stagger per row
#define STRIDE_H 2056  // 2048 + 8 bf16 pad

typedef __attribute__((ext_vector_type(16))) __bf16 v16bf;
typedef __attribute__((ext_vector_type(8)))  float  v8f;

union Frag {
  uint4 q[2];
  v16bf v;
};

__device__ __forceinline__ unsigned short f2bf(float f) {
  return __builtin_bit_cast(unsigned short, (__bf16)f);
}

// ---- weight convert + transpose: src [E][R][C] f32 -> dst [E][C][R] bf16 ----
__global__ __launch_bounds__(256) void cvt_t_kernel(const float* __restrict__ src,
                                                    unsigned short* __restrict__ dst,
                                                    int R, int C) {
  long idx = (long)blockIdx.x * 256 + threadIdx.x;
  long total = (long)E_NUM * R * C;
  if (idx >= total) return;
  int c = (int)(idx % C);
  long t = idx / C;
  int r = (int)(t % R);
  int e = (int)(t / R);
  dst[((long)e * C + c) * R + r] = f2bf(src[idx]);
}

__global__ void init_kernel(int* counts, float* wsum) {
  int t = threadIdx.x;
  if (t < E_NUM) { counts[t] = 0; wsum[t] = 0.0f; }
}

__global__ __launch_bounds__(256) void zero_f(float* p, long n) {
  long i = (long)blockIdx.x * 256 + threadIdx.x;
  if (i < n) p[i] = 0.0f;
}

// ---- router: logits -> softmax -> top2; counts + weight sums for aux ----
__global__ __launch_bounds__(256) void router_kernel(const float* __restrict__ X,
                                                     const float* __restrict__ RW,
                                                     int* __restrict__ sel,
                                                     float* __restrict__ selw,
                                                     int* __restrict__ counts,
                                                     float* __restrict__ wsum) {
  __shared__ float rw[D_DIM * E_NUM];  // 16 KB
  int tid = threadIdx.x;
  for (int i = tid; i < D_DIM * E_NUM; i += 256) rw[i] = RW[i];
  __syncthreads();

  int tok = blockIdx.x * 256 + tid;
  const float* xr = X + (size_t)tok * D_DIM;
  float acc[E_NUM] = {};
  for (int k = 0; k < D_DIM; ++k) {
    float xv = xr[k];
#pragma unroll
    for (int e = 0; e < E_NUM; ++e) acc[e] += xv * rw[k * E_NUM + e];
  }
  float m = acc[0];
#pragma unroll
  for (int e = 1; e < E_NUM; ++e) m = fmaxf(m, acc[e]);
  float p[E_NUM];
  float s = 0.0f;
#pragma unroll
  for (int e = 0; e < E_NUM; ++e) { p[e] = __expf(acc[e] - m); s += p[e]; }
  float inv = 1.0f / s;
#pragma unroll
  for (int e = 0; e < E_NUM; ++e) p[e] *= inv;

  int i0 = 0; float p0 = p[0];
#pragma unroll
  for (int e = 1; e < E_NUM; ++e) if (p[e] > p0) { p0 = p[e]; i0 = e; }
  int i1 = 0; float p1 = -1.0f;
#pragma unroll
  for (int e = 0; e < E_NUM; ++e) if (e != i0 && p[e] > p1) { p1 = p[e]; i1 = e; }

  sel[tok * 2 + 0] = i0; sel[tok * 2 + 1] = i1;
  selw[tok * 2 + 0] = p0; selw[tok * 2 + 1] = p1;
  atomicAdd(&counts[i0], 1); atomicAdd(&counts[i1], 1);
  atomicAdd(&wsum[i0], p0);  atomicAdd(&wsum[i1], p1);
}

// ---- tiny: exclusive scan over 8 experts + aux loss ----
__global__ void scan_aux_kernel(const int* counts, const float* wsum,
                                int* offsets, int* cursor, float* out_aux) {
  if (threadIdx.x == 0) {
    int accu = 0;
    float a = 0.0f;
    for (int e = 0; e < E_NUM; ++e) {
      offsets[e] = accu; accu += counts[e]; cursor[e] = 0;
      a += ((float)counts[e] / (float)N_TOK) * (wsum[e] / (float)N_TOK);
    }
    *out_aux = (float)E_NUM * a;
  }
}

// ---- scatter tokens into per-expert lists ----
__global__ __launch_bounds__(256) void assign_kernel(const int* __restrict__ sel,
                                                     const float* __restrict__ selw,
                                                     const int* __restrict__ offsets,
                                                     int* __restrict__ cursor,
                                                     int* __restrict__ tokl,
                                                     float* __restrict__ wl) {
  int tok = blockIdx.x * 256 + threadIdx.x;
  if (tok >= N_TOK) return;
#pragma unroll
  for (int j = 0; j < 2; ++j) {
    int e = sel[tok * 2 + j];
    int pos = atomicAdd(&cursor[e], 1);
    tokl[offsets[e] + pos] = tok;
    wl[offsets[e] + pos] = selw[tok * 2 + j];
  }
}

// ---- fused expert FFN over one 32-token tile: 2x2 register-blocked WMMA ----
__global__ __launch_bounds__(256) void ffn_kernel(
    const float* __restrict__ X,
    const unsigned short* __restrict__ w1t,  // [E][H][D] bf16 (pre-transposed)
    const unsigned short* __restrict__ w2t,  // [E][D][H] bf16 (pre-transposed)
    const float* __restrict__ b1,            // [E][H]
    const float* __restrict__ b2,            // [E][D]
    const int* __restrict__ counts,
    const int* __restrict__ offsets,
    const int* __restrict__ tokl,
    const float* __restrict__ wl,
    float* __restrict__ out) {
  int e = blockIdx.y;
  int cnt = counts[e];
  int tileBase = blockIdx.x * TILE_M;
  if (tileBase >= cnt) return;  // uniform: EXEC stays all-ones for WMMA

  extern __shared__ char smem[];
  unsigned short* A_sh = (unsigned short*)smem;            // 32 x STRIDE_A bf16
  unsigned short* H_sh = A_sh + TILE_M * STRIDE_A;         // 32 x STRIDE_H bf16
  int*   s_tok = (int*)(H_sh + TILE_M * STRIDE_H);
  float* s_w   = (float*)(s_tok + TILE_M);

  int tid = threadIdx.x;
  int off = offsets[e];
  if (tid < TILE_M) {
    int r = tileBase + tid;
    int tok = 0; float w = 0.0f;
    if (r < cnt) { tok = tokl[off + r]; w = wl[off + r]; }
    s_tok[tid] = tok; s_w[tid] = w;
  }
  __syncthreads();

  // gather 32 token rows of X, convert fp32 -> bf16 into LDS
#pragma unroll 4
  for (int i = 0; i < 64; ++i) {
    int flat = i * 256 + tid;          // 0..16383
    int row = flat >> 9;
    int col = flat & 511;
    float v = X[(size_t)s_tok[row] * D_DIM + col];
    A_sh[row * STRIDE_A + col] = f2bf(v);
  }
  __syncthreads();

  int wave = tid >> 5;
  int lane = tid & 31;
  int nl   = lane & 15;   // A row index / B,C column index for this lane
  int hi   = lane >> 4;   // half-wave select

  // ---- GEMM1: [32 x 512] x [512 x 2048] -> relu -> H_sh bf16 ----
  // 2x2 blocking: A frag shared across N-pair, B frag shared across M-pair.
  for (int np = 0; np < 8; ++np) {
    int n0 = wave * 256 + np * 32 + nl;
    int n1 = n0 + 16;
    v8f acc00 = {}, acc01 = {}, acc10 = {}, acc11 = {};
    const unsigned short* bcol0 = w1t + ((size_t)(e * H_DIM + n0)) * D_DIM;
    const unsigned short* bcol1 = w1t + ((size_t)(e * H_DIM + n1)) * D_DIM;
#pragma unroll 2
    for (int k0 = 0; k0 < D_DIM; k0 += 32) {
      Frag a0, a1, b0, b1f;
      // A 16-bit layout: elems 0-15 -> K=k0+hi*16.. contiguous
      a0.q[0] = *(const uint4*)&A_sh[nl * STRIDE_A + k0 + hi * 16];
      a0.q[1] = *(const uint4*)&A_sh[nl * STRIDE_A + k0 + 8 + hi * 16];
      a1.q[0] = *(const uint4*)&A_sh[(16 + nl) * STRIDE_A + k0 + hi * 16];
      a1.q[1] = *(const uint4*)&A_sh[(16 + nl) * STRIDE_A + k0 + 8 + hi * 16];
      // B 16-bit layout: lane=col, K = k0 + hi*16 + j contiguous (pre-transposed)
      const uint4* bp0 = (const uint4*)(bcol0 + k0 + hi * 16);
      const uint4* bp1 = (const uint4*)(bcol1 + k0 + hi * 16);
      b0.q[0] = bp0[0]; b0.q[1] = bp0[1];
      b1f.q[0] = bp1[0]; b1f.q[1] = bp1[1];
      acc00 = __builtin_amdgcn_wmma_f32_16x16x32_bf16(false, a0.v, false, b0.v,
                                                      (short)0, acc00, false, false);
      acc01 = __builtin_amdgcn_wmma_f32_16x16x32_bf16(false, a0.v, false, b1f.v,
                                                      (short)0, acc01, false, false);
      acc10 = __builtin_amdgcn_wmma_f32_16x16x32_bf16(false, a1.v, false, b0.v,
                                                      (short)0, acc10, false, false);
      acc11 = __builtin_amdgcn_wmma_f32_16x16x32_bf16(false, a1.v, false, b1f.v,
                                                      (short)0, acc11, false, false);
    }
    float bias0 = b1[e * H_DIM + n0];
    float bias1 = b1[e * H_DIM + n1];
#pragma unroll
    for (int v = 0; v < 8; ++v) {
      int r0 = v + hi * 8;       // C/D layout: M = v + 8*hi, N = nl
      int r1 = r0 + 16;
      float h;
      h = acc00[v] + bias0; h = h > 0.0f ? h : 0.0f;
      H_sh[r0 * STRIDE_H + n0] = f2bf(h);
      h = acc01[v] + bias1; h = h > 0.0f ? h : 0.0f;
      H_sh[r0 * STRIDE_H + n1] = f2bf(h);
      h = acc10[v] + bias0; h = h > 0.0f ? h : 0.0f;
      H_sh[r1 * STRIDE_H + n0] = f2bf(h);
      h = acc11[v] + bias1; h = h > 0.0f ? h : 0.0f;
      H_sh[r1 * STRIDE_H + n1] = f2bf(h);
    }
  }
  __syncthreads();

  // ---- GEMM2: [32 x 2048] x [2048 x 512] -> weighted atomic scatter ----
  for (int np = 0; np < 2; ++np) {
    int n0 = wave * 64 + np * 32 + nl;
    int n1 = n0 + 16;
    v8f acc00 = {}, acc01 = {}, acc10 = {}, acc11 = {};
    const unsigned short* bcol0 = w2t + ((size_t)(e * D_DIM + n0)) * H_DIM;
    const unsigned short* bcol1 = w2t + ((size_t)(e * D_DIM + n1)) * H_DIM;
#pragma unroll 2
    for (int k0 = 0; k0 < H_DIM; k0 += 32) {
      Frag a0, a1, b0, b1f;
      a0.q[0] = *(const uint4*)&H_sh[nl * STRIDE_H + k0 + hi * 16];
      a0.q[1] = *(const uint4*)&H_sh[nl * STRIDE_H + k0 + 8 + hi * 16];
      a1.q[0] = *(const uint4*)&H_sh[(16 + nl) * STRIDE_H + k0 + hi * 16];
      a1.q[1] = *(const uint4*)&H_sh[(16 + nl) * STRIDE_H + k0 + 8 + hi * 16];
      const uint4* bp0 = (const uint4*)(bcol0 + k0 + hi * 16);
      const uint4* bp1 = (const uint4*)(bcol1 + k0 + hi * 16);
      b0.q[0] = bp0[0]; b0.q[1] = bp0[1];
      b1f.q[0] = bp1[0]; b1f.q[1] = bp1[1];
      acc00 = __builtin_amdgcn_wmma_f32_16x16x32_bf16(false, a0.v, false, b0.v,
                                                      (short)0, acc00, false, false);
      acc01 = __builtin_amdgcn_wmma_f32_16x16x32_bf16(false, a0.v, false, b1f.v,
                                                      (short)0, acc01, false, false);
      acc10 = __builtin_amdgcn_wmma_f32_16x16x32_bf16(false, a1.v, false, b0.v,
                                                      (short)0, acc10, false, false);
      acc11 = __builtin_amdgcn_wmma_f32_16x16x32_bf16(false, a1.v, false, b1f.v,
                                                      (short)0, acc11, false, false);
    }
    float bias0 = b2[e * D_DIM + n0];
    float bias1 = b2[e * D_DIM + n1];
#pragma unroll
    for (int v = 0; v < 8; ++v) {
      int m0 = v + hi * 8;
      int m1 = m0 + 16;
      if (tileBase + m0 < cnt) {
        float y0 = acc00[v] + bias0;
        float y1 = acc01[v] + bias1;
        atomicAdd(&out[(size_t)s_tok[m0] * D_DIM + n0], s_w[m0] * y0);
        atomicAdd(&out[(size_t)s_tok[m0] * D_DIM + n1], s_w[m0] * y1);
      }
      if (tileBase + m1 < cnt) {
        float y0 = acc10[v] + bias0;
        float y1 = acc11[v] + bias1;
        atomicAdd(&out[(size_t)s_tok[m1] * D_DIM + n0], s_w[m1] * y0);
        atomicAdd(&out[(size_t)s_tok[m1] * D_DIM + n1], s_w[m1] * y1);
      }
    }
  }
}

extern "C" void kernel_launch(void* const* d_in, const int* in_sizes, int n_in,
                              void* d_out, int out_size, void* d_ws, size_t ws_size,
                              hipStream_t stream) {
  const float* X  = (const float*)d_in[0];
  const float* RW = (const float*)d_in[1];
  const float* w1 = (const float*)d_in[2];
  const float* b1 = (const float*)d_in[3];
  const float* w2 = (const float*)d_in[4];
  const float* b2 = (const float*)d_in[5];
  float* out = (float*)d_out;

  // workspace layout
  char* ws = (char*)d_ws;
  const size_t W_ELEMS = (size_t)E_NUM * D_DIM * H_DIM;  // 8,388,608
  unsigned short* w1t = (unsigned short*)(ws);                       // 16 MB
  unsigned short* w2t = (unsigned short*)(ws + W_ELEMS * 2);         // 16 MB
  char* rp = ws + W_ELEMS * 4;
  int*   sel    = (int*)(rp);                    // 2N ints
  float* selw   = (float*)(rp + 65536);          // 2N floats
  int*   tokl   = (int*)(rp + 2 * 65536);        // 2N ints
  float* wl     = (float*)(rp + 3 * 65536);      // 2N floats
  int*   counts = (int*)(rp + 4 * 65536);
  float* wsum   = (float*)(rp + 4 * 65536 + 256);
  int*   offs   = (int*)(rp + 4 * 65536 + 512);
  int*   cursor = (int*)(rp + 4 * 65536 + 768);

  // 1) pre-convert weights fp32 -> bf16 with transpose (L2-resident)
  int wblocks = (int)((W_ELEMS + 255) / 256);
  cvt_t_kernel<<<wblocks, 256, 0, stream>>>(w1, w1t, D_DIM, H_DIM);  // -> [E][H][D]
  cvt_t_kernel<<<wblocks, 256, 0, stream>>>(w2, w2t, H_DIM, D_DIM);  // -> [E][D][H]

  // 2) init routing state + zero output (atomic accumulate target)
  init_kernel<<<1, 32, 0, stream>>>(counts, wsum);
  zero_f<<<(out_size + 255) / 256, 256, 0, stream>>>(out, (long)out_size);

  // 3) router -> 4) scan+aux -> 5) scatter token lists
  router_kernel<<<N_TOK / 256, 256, 0, stream>>>(X, RW, sel, selw, counts, wsum);
  scan_aux_kernel<<<1, 32, 0, stream>>>(counts, wsum, offs, cursor,
                                        out + (size_t)N_TOK * D_DIM);
  assign_kernel<<<N_TOK / 256, 256, 0, stream>>>(sel, selw, offs, cursor, tokl, wl);

  // 6) grouped expert FFN: grid = (max tiles per expert, experts)
  dim3 grid(N_TOK / TILE_M, E_NUM);  // max tokens per expert = N_TOK
  size_t smem = (size_t)(TILE_M * STRIDE_A + TILE_M * STRIDE_H) * 2 + TILE_M * 8;
  ffn_kernel<<<grid, 256, smem, stream>>>(X, w1t, w2t, b1, b2,
                                          counts, offs, tokl, wl, out);
}
